// GraphTransEncoderLayer_85186381349287
// MI455X (gfx1250) — matrix operand order repaired and checked by reading, not compiled, not run on hardware
//
#include <hip/hip_runtime.h>
#include <hip/hip_bf16.h>
#include <cstdint>

// ---------------------------------------------------------------------------
// GATv2 Graph-Transformer encoder layer for gfx1250 (MI455X, wave32, WMMA).
// N=50000 nodes, E=320000 edges, F=256 feats, H=8 heads, C=32 per-head.
//  - GEMMs via v_wmma_f32_16x16x32_bf16 (f32 accumulate)
//  - weights kept TRANSPOSED in bf16 so B fragments are contiguous b128 loads
//  - edge projection fused with attention logits (no 327MB intermediate)
//  - scatter-softmax: atomic float max -> exp/atomic-add den -> unnormalized
//    scatter-add; the 1/den normalization is folded into the final LN kernel.
// ---------------------------------------------------------------------------

#define KN 50000
#define KE 320000
#define KF 256
#define KH 8

typedef __attribute__((ext_vector_type(16))) __bf16 v16bf;
typedef __attribute__((ext_vector_type(8)))  __bf16 v8bf;
typedef __attribute__((ext_vector_type(8)))  float  v8f;

// ---- helpers --------------------------------------------------------------

__device__ __forceinline__ void atomicMaxF(float* addr, float val) {
  // sortable-bits float max: positives ordered as int, negatives reverse as uint
  if (val >= 0.0f) atomicMax((int*)addr, __float_as_int(val));
  else             atomicMin((unsigned int*)addr, __float_as_uint(val));
}

// Build a 16x32 bf16 A fragment from 4x float4 (row-major fp32 source).
// Element i -> k = (i/8)*16 + half*8 + (i%8); caller passes pa = row + k0 + half*8.
__device__ __forceinline__ v16bf load_a_frag_f32(const float* __restrict__ pa) {
  const float4 f0 = ((const float4*)pa)[0];
  const float4 f1 = ((const float4*)pa)[1];
  const float4 f2 = ((const float4*)(pa + 16))[0];
  const float4 f3 = ((const float4*)(pa + 16))[1];
  v16bf a;
  a[0]  = (__bf16)f0.x; a[1]  = (__bf16)f0.y; a[2]  = (__bf16)f0.z; a[3]  = (__bf16)f0.w;
  a[4]  = (__bf16)f1.x; a[5]  = (__bf16)f1.y; a[6]  = (__bf16)f1.z; a[7]  = (__bf16)f1.w;
  a[8]  = (__bf16)f2.x; a[9]  = (__bf16)f2.y; a[10] = (__bf16)f2.z; a[11] = (__bf16)f2.w;
  a[12] = (__bf16)f3.x; a[13] = (__bf16)f3.y; a[14] = (__bf16)f3.z; a[15] = (__bf16)f3.w;
  return a;
}

// Build a 16x32 bf16 A fragment from bf16 source (two v8bf vector loads).
__device__ __forceinline__ v16bf load_a_frag_bf16(const __bf16* __restrict__ pa) {
  const v8bf lo = ((const v8bf*)pa)[0];        // k = half*8 + 0..7
  const v8bf hi = ((const v8bf*)(pa + 16))[0]; // k = 16 + half*8 + 0..7
  v16bf a;
#pragma unroll
  for (int i = 0; i < 8; ++i) { a[i] = lo[i]; a[8 + i] = hi[i]; }
  return a;
}

// Build a 32x16 bf16 B fragment from TRANSPOSED weights Wt[n][k]:
// per-lane column n fixed, k = half*8 + i and 16 + half*8 + i -> two b128 loads.
__device__ __forceinline__ v16bf load_b_frag(const __bf16* __restrict__ pb) {
  const v8bf b0 = ((const v8bf*)pb)[0];
  const v8bf b1 = ((const v8bf*)(pb + 16))[0];
  v16bf b;
#pragma unroll
  for (int i = 0; i < 8; ++i) { b[i] = b0[i]; b[8 + i] = b1[i]; }
  return b;
}

// ---- conversions ----------------------------------------------------------

__global__ void k_cvt_bf16(const float* __restrict__ src,
                           __bf16* __restrict__ dst, int n) {
  int i = blockIdx.x * blockDim.x + threadIdx.x;
  if (i < n) dst[i] = (__bf16)src[i];
}

// W [K=256, N=256] fp32 row-major -> Wt [N=256, K=256] bf16 row-major
__global__ void k_cvt_transpose(const float* __restrict__ W,
                                __bf16* __restrict__ Wt) {
  int i = blockIdx.x * blockDim.x + threadIdx.x;   // 65536
  int k = i >> 8, n = i & 255;
  Wt[(size_t)n * KF + k] = (__bf16)W[(size_t)k * KF + n];
}

// ---- init: zero output accumulator, -inf amax, zero den -------------------

__global__ void k_init(float* __restrict__ out, float* __restrict__ amax,
                       float* __restrict__ den) {
  int i = blockIdx.x * blockDim.x + threadIdx.x;
  if (i < KN * KF) out[i] = 0.0f;
  if (i < KN * KH) { amax[i] = -3.402823466e38f; den[i] = 0.0f; }
}

// ---- node GEMM: C[M,256] = A_bf16[M,256] @ W[256,256] + bias --------------
// one wave per 16x16 output tile; 8 WMMA k-steps of 32; B from transposed Wt.

__global__ void k_gemm_nodes(const __bf16* __restrict__ A,
                             const __bf16* __restrict__ Wt,
                             const float* __restrict__ bias,
                             float* __restrict__ C) {
  const int lane = threadIdx.x;        // 0..31 (wave32)
  const int half = lane >> 4;
  const int mr   = lane & 15;
  const int m0   = blockIdx.x * 16;
  const int n0   = blockIdx.y * 16;

  const __bf16* arow = A  + (size_t)(m0 + mr) * KF + half * 8;
  const __bf16* brow = Wt + (size_t)(n0 + mr) * KF + half * 8;

  v8f acc = {};
#pragma unroll
  for (int ks = 0; ks < 8; ++ks) {
    const int k0 = ks * 32;
    const v16bf a = load_a_frag_bf16(arow + k0);
    const v16bf b = load_b_frag(brow + k0);
    acc = __builtin_amdgcn_wmma_f32_16x16x32_bf16(
        false, a, false, b, (short)0, acc, false, false);
  }
  const float bv = bias[n0 + mr];
#pragma unroll
  for (int r = 0; r < 8; ++r) {
    const int row = r + half * 8;
    C[(size_t)(m0 + row) * KF + n0 + mr] = acc[r] + bv;
  }
}

// ---- fused edge kernel ----------------------------------------------------
// One wave handles 16 edges. For each 16-col tile: WMMA-project edge_attr
// through W_e, add bias + gathered x_l[src] + x_r[dst], leaky-ReLU, dot with
// att, lane-reduce over the 16 columns, accumulate per head (2 tiles/head).
// Writes alpha logits [E,H] and atomic-max into amax[N,H].

__global__ void k_edge_alpha(const float* __restrict__ ea,
                             const __bf16* __restrict__ Wet,
                             const float* __restrict__ be,
                             const int* __restrict__ ei,
                             const float* __restrict__ xl,
                             const float* __restrict__ xr,
                             const float* __restrict__ att,
                             float* __restrict__ alpha,
                             float* __restrict__ amax) {
  __shared__ int s_src[16], s_dst[16];
  const int e0   = blockIdx.x * 16;
  const int lane = threadIdx.x;
  if (lane < 16) {
    s_src[lane] = ei[e0 + lane];        // edge_index[0, :]
    s_dst[lane] = ei[KE + e0 + lane];   // edge_index[1, :]
  }
  __syncthreads();

  const int half = lane >> 4;
  const int mr   = lane & 15;
  const float* arow = ea + (size_t)(e0 + mr) * KF + half * 8;
  float asum[8];

  for (int nt = 0; nt < 16; ++nt) {
    const __bf16* brow = Wet + (size_t)(nt * 16 + mr) * KF + half * 8;
    v8f acc = {};
#pragma unroll
    for (int ks = 0; ks < 8; ++ks) {
      const int k0 = ks * 32;
      const v16bf a = load_a_frag_f32(arow + k0);
      const v16bf b = load_b_frag(brow + k0);
      acc = __builtin_amdgcn_wmma_f32_16x16x32_bf16(
          false, a, false, b, (short)0, acc, false, false);
    }

    const int   gcol = nt * 16 + mr;   // global feature column 0..255
    const float attv = att[gcol];      // att is [H,C] flat = 256
    const float bev  = be[gcol];

#pragma unroll
    for (int r = 0; r < 8; ++r) {
      const int row = r + half * 8;    // edge within the 16-edge tile
      float v = acc[r] + bev
              + xl[(size_t)s_src[row] * KF + gcol]
              + xr[(size_t)s_dst[row] * KF + gcol];
      v = v > 0.0f ? v : 0.1f * v;     // leaky_relu(0.1)
      float c = v * attv;
      // reduce over the 16 columns (lanes within each 16-lane half)
#pragma unroll
      for (int off = 1; off < 16; off <<= 1) c += __shfl_xor(c, off, 32);
      if ((nt & 1) == 0) asum[r] = c; else asum[r] += c;
    }

    if (nt & 1) {                      // head complete (2 tiles per head)
      const int h = nt >> 1;
#pragma unroll
      for (int r = 0; r < 8; ++r) {
        if (mr == r) {                 // lanes r and 16+r write rows r, r+8
          const int row  = r + half * 8;
          const int eidx = e0 + row;
          alpha[(size_t)eidx * KH + h] = asum[r];
          atomicMaxF(&amax[(size_t)s_dst[row] * KH + h], asum[r]);
        }
      }
    }
  }
}

// ---- exp + denominator pass ----------------------------------------------

__global__ void k_edge_exp(const int* __restrict__ ei,
                           float* __restrict__ alpha,
                           const float* __restrict__ amax,
                           float* __restrict__ den) {
  int i = blockIdx.x * blockDim.x + threadIdx.x;
  if (i >= KE * KH) return;
  const int e = i >> 3, h = i & 7;
  const int dst = ei[KE + e];
  const float ex = expf(alpha[i] - amax[(size_t)dst * KH + h]);
  alpha[i] = ex;
  atomicAdd(&den[(size_t)dst * KH + h], ex);
}

// ---- UNNORMALIZED weighted scatter-add of messages ------------------------
// out[dst,f] += ex(e,h) * x_l[src,f]; division by den deferred to LN kernel.

__global__ void k_message(const int* __restrict__ ei,
                          const float* __restrict__ alpha,
                          const float* __restrict__ xl,
                          float* __restrict__ out) {
  int i = blockIdx.x * blockDim.x + threadIdx.x;   // E*256 = 81.92M < 2^31
  if (i >= KE * KF) return;
  const int e = i >> 8, f = i & 255, h = f >> 5;
  const int src = ei[e], dst = ei[KE + e];
  const float w = alpha[(size_t)e * KH + h];
  atomicAdd(&out[(size_t)dst * KF + f], xl[(size_t)src * KF + f] * w);
}

// ---- normalize + residual + LayerNorm + exact GELU (one block per node) ---

__global__ void k_ln_gelu(const float* __restrict__ x,
                          const float* __restrict__ den,
                          const float* __restrict__ bias_out,
                          const float* __restrict__ gamma,
                          const float* __restrict__ beta,
                          float* __restrict__ out) {
  const int n = blockIdx.x, f = threadIdx.x;
  const size_t idx = (size_t)n * KF + f;
  const float dnv = den[(size_t)n * KH + (f >> 5)] + 1e-16f;
  const float h = x[idx] + out[idx] / dnv + bias_out[f];

  float s = h, s2 = h * h;
#pragma unroll
  for (int off = 16; off; off >>= 1) {
    s  += __shfl_down(s,  off, 32);
    s2 += __shfl_down(s2, off, 32);
  }
  __shared__ float sh[8], sh2[8];
  if ((f & 31) == 0) { sh[f >> 5] = s; sh2[f >> 5] = s2; }
  __syncthreads();
  float ts = 0.0f, ts2 = 0.0f;
#pragma unroll
  for (int i = 0; i < 8; ++i) { ts += sh[i]; ts2 += sh2[i]; }

  const float mu  = ts * (1.0f / KF);
  const float var = ts2 * (1.0f / KF) - mu * mu;
  const float y   = (h - mu) * rsqrtf(var + 1e-5f) * gamma[f] + beta[f];
  out[idx] = 0.5f * y * (1.0f + erff(y * 0.70710678118654752f));
}

// ---------------------------------------------------------------------------

extern "C" void kernel_launch(void* const* d_in, const int* in_sizes, int n_in,
                              void* d_out, int out_size, void* d_ws,
                              size_t ws_size, hipStream_t stream) {
  const float* x        = (const float*)d_in[0];
  const int*   ei       = (const int*)  d_in[1];   // [2,E] src;dst
  const float* ea       = (const float*)d_in[2];
  const float* W_l      = (const float*)d_in[3];
  const float* b_l      = (const float*)d_in[4];
  const float* W_r      = (const float*)d_in[5];
  const float* b_r      = (const float*)d_in[6];
  const float* W_e      = (const float*)d_in[7];
  const float* b_e      = (const float*)d_in[8];
  const float* att      = (const float*)d_in[9];
  const float* bias_out = (const float*)d_in[10];
  const float* gamma    = (const float*)d_in[11];
  const float* beta     = (const float*)d_in[12];
  float* out = (float*)d_out;

  // workspace carve-up (~142 MB)
  char* ws = (char*)d_ws;
  size_t off = 0;
  auto carve = [&](size_t bytes) {
    void* p = ws + off;
    off = (off + bytes + 255) & ~(size_t)255;
    return p;
  };
  __bf16* xb    = (__bf16*)carve((size_t)KN * KF * 2);
  __bf16* Wlt   = (__bf16*)carve((size_t)KF * KF * 2);  // transposed bf16
  __bf16* Wrt   = (__bf16*)carve((size_t)KF * KF * 2);
  __bf16* Wet   = (__bf16*)carve((size_t)KF * KF * 2);
  float*  x_l   = (float*) carve((size_t)KN * KF * 4);
  float*  x_r   = (float*) carve((size_t)KN * KF * 4);
  float*  alpha = (float*) carve((size_t)KE * KH * 4);
  float*  amax  = (float*) carve((size_t)KN * KH * 4);
  float*  den   = (float*) carve((size_t)KN * KH * 4);
  (void)ws_size; (void)n_in; (void)in_sizes; (void)out_size;

  // 1) conversions + init
  k_cvt_bf16<<<(KN * KF + 255) / 256, 256, 0, stream>>>(x, xb, KN * KF);
  k_cvt_transpose<<<(KF * KF) / 256, 256, 0, stream>>>(W_l, Wlt);
  k_cvt_transpose<<<(KF * KF) / 256, 256, 0, stream>>>(W_r, Wrt);
  k_cvt_transpose<<<(KF * KF) / 256, 256, 0, stream>>>(W_e, Wet);
  k_init<<<(KN * KF + 255) / 256, 256, 0, stream>>>(out, amax, den);

  // 2) node GEMMs via WMMA: x_l = x@W_l+b_l, x_r = x@W_r+b_r
  dim3 ggrid(KN / 16, KF / 16);
  k_gemm_nodes<<<ggrid, 32, 0, stream>>>(xb, Wlt, b_l, x_l);
  k_gemm_nodes<<<ggrid, 32, 0, stream>>>(xb, Wrt, b_r, x_r);

  // 3) fused edge projection + attention logits + segment max
  k_edge_alpha<<<KE / 16, 32, 0, stream>>>(ea, Wet, b_e, ei, x_l, x_r, att,
                                           alpha, amax);

  // 4) softmax exp + denominators
  k_edge_exp<<<(KE * KH + 255) / 256, 256, 0, stream>>>(ei, alpha, amax, den);

  // 5) unnormalized weighted scatter-add of messages
  k_message<<<(KE * KF) / 256, 256, 0, stream>>>(ei, alpha, x_l, out);

  // 6) normalize + residual + LayerNorm + GELU
  k_ln_gelu<<<KN, KF, 0, stream>>>(x, den, bias_out, gamma, beta, out);
}
